// MultiHeadAttention_59803124629980
// MI455X (gfx1250) — matrix operand order repaired and checked by reading, not compiled, not run on hardware
//
#include <hip/hip_runtime.h>

// ---------- types ----------
typedef __bf16 bf16;
typedef __attribute__((ext_vector_type(16))) __bf16 v16bf;
typedef __attribute__((ext_vector_type(8)))  __bf16 v8bf;
typedef __attribute__((ext_vector_type(8)))  float  v8f;
typedef __attribute__((ext_vector_type(4)))  unsigned int v4u;
typedef __attribute__((ext_vector_type(8)))  int v8i_t;
typedef __attribute__((ext_vector_type(4)))  int v4i_t;

typedef __attribute__((address_space(1))) v4i_t gv4i;   // global int4 (async src)
typedef __attribute__((address_space(3))) v4i_t lv4i;   // LDS int4 (async dst)
typedef __attribute__((address_space(3))) __bf16 lds_bf16;

// ---------- feature probes (degrade gracefully if builtin absent) ----------
#if defined(__HIP_DEVICE_COMPILE__) && __has_builtin(__builtin_amdgcn_global_load_async_to_lds_b128)
#define HAVE_ASYNC 1
#else
#define HAVE_ASYNC 0
#endif
#if defined(__HIP_DEVICE_COMPILE__) && __has_builtin(__builtin_amdgcn_tensor_load_to_lds)
#define HAVE_TDM 1
#else
#define HAVE_TDM 0
#endif

__device__ inline void wait_async0() {
#if __has_builtin(__builtin_amdgcn_s_wait_asynccnt)
  __builtin_amdgcn_s_wait_asynccnt(0);
#else
  asm volatile("s_wait_asynccnt 0" ::: "memory");
#endif
}
__device__ inline void wait_tensor0() {
#if __has_builtin(__builtin_amdgcn_s_wait_tensorcnt)
  __builtin_amdgcn_s_wait_tensorcnt(0);
#else
  asm volatile("s_wait_tensorcnt 0" ::: "memory");
#endif
}

// ---------- constants ----------
#define NBATCH 4
#define SEQ    2048
#define EMB    1024
#define NHEAD  16
#define DHEAD  64
#define NTOK   (NBATCH * SEQ)     // 8192

// ---------- WMMA wrapper ----------
__device__ inline v8f wmma_bf16(v16bf a, v16bf b, v8f c) {
  return __builtin_amdgcn_wmma_f32_16x16x32_bf16(
      false, a, false, b, (short)0, c, false, false);
}

// A fragment 16x32: lanes 0-15 row M=lane K={0..7,16..23}; lanes 16-31 K={8..15,24..31}
__device__ inline v16bf load_A_frag(const bf16* src, int ld) {
  int lane = threadIdx.x & 31;
  int row  = lane & 15;
  int hi   = lane >> 4;
  const bf16* p = src + row * ld + hi * 8;
  union { v16bf v; v8bf h[2]; } u;
  u.h[0] = *(const v8bf*)(p);
  u.h[1] = *(const v8bf*)(p + 16);
  return u.v;
}

// B fragment 32(K)x16(N) from [n][k] rows: lanes 0-15 K={0..7,16..23}; lanes 16-31 K={8..15,24..31}
__device__ inline v16bf load_B_frag(const bf16* src, int ld) {
  int lane = threadIdx.x & 31;
  const bf16* p = src + (lane & 15) * ld + (lane >> 4) * 8;
  union { v16bf v; v8bf h[2]; } u;
  u.h[0] = *(const v8bf*)(p);
  u.h[1] = *(const v8bf*)(p + 16);
  return u.v;
}

__device__ inline float red_max16(float x) {
  x = fmaxf(x, __shfl_xor(x, 1, 32));
  x = fmaxf(x, __shfl_xor(x, 2, 32));
  x = fmaxf(x, __shfl_xor(x, 4, 32));
  x = fmaxf(x, __shfl_xor(x, 8, 32));
  return x;
}
__device__ inline float red_sum16(float x) {
  x += __shfl_xor(x, 1, 32);
  x += __shfl_xor(x, 2, 32);
  x += __shfl_xor(x, 4, 32);
  x += __shfl_xor(x, 8, 32);
  return x;
}

// =====================================================================
// Kernel: causal flash attention (FIRST in file so disasm snippet shows it).
// 64-key blocks: K tile via TDM, V^T tile via async global->LDS b128.
// grid=(32,64), block=128 (4 waves, each owns a 16-row Q strip).
// __launch_bounds__(128,1): relax VGPR budget -> no scratch spills.
// =====================================================================
__global__ void __launch_bounds__(128, 1)
attn_kernel(const bf16* __restrict__ Qw,
            const bf16* __restrict__ Kw,
            const bf16* __restrict__ Vw,   // transposed (N,H,D,L)
            bf16* __restrict__ ctx) {
  __shared__ alignas(32) bf16 Ks[64 * 64];     // [key][d]
  __shared__ alignas(32) bf16 Vts[64 * 64];    // [d][key]
  __shared__ alignas(32) bf16 Ps[4 * 16 * 64]; // per-wave P staging

  const int tid  = threadIdx.x;
  const int w    = tid >> 5;
  const int lane = tid & 31;
  const int nloc = lane & 15;
  const int half8 = (lane >> 4) * 8;

  const int bh   = blockIdx.y;
  const int b    = bh >> 4;
  const int h    = bh & 15;
  const int q0b  = blockIdx.x * 64;
  const int qpos = q0b + w * 16;

  const bf16* Qb = Qw + (long)bh * SEQ * DHEAD;
  const bf16* Kb = Kw + (long)bh * SEQ * DHEAD;
  const bf16* Vb = Vw + (long)bh * DHEAD * SEQ;   // rows d, cols l

  v16bf qa0 = load_A_frag(Qb + qpos * DHEAD, DHEAD);
  v16bf qa1 = load_A_frag(Qb + qpos * DHEAD + 32, DHEAD);

  v8f o[4], mrun, lrun;
  #pragma unroll
  for (int i = 0; i < 8; ++i) { mrun[i] = -1e30f; lrun[i] = 0.f; }
  #pragma unroll
  for (int j = 0; j < 4; ++j)
    #pragma unroll
    for (int i = 0; i < 8; ++i) o[j][i] = 0.f;

#if HAVE_TDM
  const unsigned ks_lds = (unsigned)(unsigned long long)(lds_bf16*)Ks;
#endif

  const int jend = q0b + 64;
  for (int j0 = 0; j0 < jend; j0 += 64) {
    __syncthreads();

    // ---- K tile: 64(key) x 64(d), contiguous rows ----
#if HAVE_TDM
    if (w == 0) {
      unsigned long long ga = (unsigned long long)(const void*)(Kb + j0 * DHEAD);
      v4u g0;
      g0.x = 1u;                                        // count=1
      g0.y = ks_lds;                                    // lds_addr
      g0.z = (unsigned)(ga & 0xFFFFFFFFu);              // global_addr lo
      g0.w = (unsigned)((ga >> 32) & 0x1FFFFFFu) | (2u << 30); // hi | type=2
      v8i_t g1;
      g1[0] = (int)(1u << 16);          // data_size = 2B
      g1[1] = (int)(64u << 16);         // tensor_dim0 = 64
      g1[2] = (int)(2048u << 16);       // tensor_dim1 low16 @ bits 95:80
      g1[3] = (int)(64u << 16);         // tile_dim0 = 64
      g1[4] = 64;                       // tile_dim1 = 64
      g1[5] = 64;                       // tensor_dim0_stride = 64
      g1[6] = 0; g1[7] = 0;
      v4i_t gz4 = {0, 0, 0, 0};
#if __clang_major__ >= 23
      v8i_t gz8 = {0, 0, 0, 0, 0, 0, 0, 0};
      __builtin_amdgcn_tensor_load_to_lds(g0, g1, gz4, gz4, gz8, 0);
#else
      __builtin_amdgcn_tensor_load_to_lds(g0, g1, gz4, gz4, 0);
#endif
    }
#elif HAVE_ASYNC
    for (int cch = tid; cch < 512; cch += 128) {        // 16B chunks
      int r = cch >> 3, cc8 = (cch & 7) * 8;
      __builtin_amdgcn_global_load_async_to_lds_b128(
          (gv4i*)(Kb + (j0 + r) * DHEAD + cc8),
          (lv4i*)(Ks + r * 64 + cc8), 0, 0);
    }
#else
    for (int i = tid; i < 64 * 64; i += 128) {
      int r = i >> 6, cc = i & 63;
      Ks[r * 64 + cc] = Kb[(j0 + r) * DHEAD + cc];
    }
#endif

    // ---- V^T tile: 64(d) x 64(key), contiguous 64-key rows ----
#if HAVE_ASYNC
    for (int cch = tid; cch < 512; cch += 128) {        // 16B chunks
      int d = cch >> 3, kk8 = (cch & 7) * 8;
      __builtin_amdgcn_global_load_async_to_lds_b128(
          (gv4i*)(Vb + d * SEQ + j0 + kk8),
          (lv4i*)(Vts + d * 64 + kk8), 0, 0);
    }
    wait_async0();
#else
    for (int i = tid; i < 64 * 64; i += 128) {
      int d = i >> 6, kk = i & 63;
      Vts[d * 64 + kk] = Vb[d * SEQ + j0 + kk];
    }
#endif
#if HAVE_TDM
    if (w == 0) wait_tensor0();
#endif
    __syncthreads();

    // ---- scores: four 16-key tiles, contract d=64 in two k-steps ----
    v8f s[4];
    #pragma unroll
    for (int jt = 0; jt < 4; ++jt) {
      v8f acc;
      #pragma unroll
      for (int i = 0; i < 8; ++i) acc[i] = 0.f;
      v16bf kb0 = load_B_frag(Ks + jt * 16 * 64, 64);
      v16bf kb1 = load_B_frag(Ks + jt * 16 * 64 + 32, 64);
      acc = wmma_bf16(qa0, kb0, acc);
      acc = wmma_bf16(qa1, kb1, acc);
      int kidx = j0 + jt * 16 + nloc;
      #pragma unroll
      for (int i = 0; i < 8; ++i) {
        float sv = acc[i] * 0.125f;                  // 1/sqrt(64)
        int qi = qpos + half8 + i;
        acc[i] = (kidx <= qi) ? sv : -1e30f;         // causal mask
      }
      s[jt] = acc;
    }

    // ---- online softmax ----
    v8f mnew, corr;
    #pragma unroll
    for (int i = 0; i < 8; ++i) {
      float t = fmaxf(fmaxf(s[0][i], s[1][i]), fmaxf(s[2][i], s[3][i]));
      t = red_max16(t);
      mnew[i] = fmaxf(mrun[i], t);
      corr[i] = __expf(mrun[i] - mnew[i]);
    }
    #pragma unroll
    for (int i = 0; i < 8; ++i) {
      #pragma unroll
      for (int jt = 0; jt < 4; ++jt) s[jt][i] = __expf(s[jt][i] - mnew[i]);
      float t = red_sum16((s[0][i] + s[1][i]) + (s[2][i] + s[3][i]));
      lrun[i] = lrun[i] * corr[i] + t;
    }
    #pragma unroll
    for (int j = 0; j < 4; ++j)
      #pragma unroll
      for (int i = 0; i < 8; ++i) o[j][i] *= corr[i];
    mrun = mnew;

    // ---- P (16x64): C-layout -> LDS -> two A fragments (wave-private) ----
    bf16* P = Ps + w * 16 * 64;
    #pragma unroll
    for (int jt = 0; jt < 4; ++jt)
      #pragma unroll
      for (int i = 0; i < 8; ++i)
        P[(half8 + i) * 64 + jt * 16 + nloc] = (bf16)s[jt][i];
    asm volatile("s_wait_dscnt 0" ::: "memory");

    v16bf pa0 = load_A_frag(P, 64);
    v16bf pa1 = load_A_frag(P + 32, 64);
    #pragma unroll
    for (int j = 0; j < 4; ++j) {
      v16bf vb0 = load_B_frag(Vts + j * 16 * 64, 64);
      v16bf vb1 = load_B_frag(Vts + j * 16 * 64 + 32, 64);
      o[j] = wmma_bf16(pa0, vb0, o[j]);
      o[j] = wmma_bf16(pa1, vb1, o[j]);
    }
  }

  // ---- epilogue: normalize, store context (N, L, E) bf16 ----
  #pragma unroll
  for (int j = 0; j < 4; ++j)
    #pragma unroll
    for (int i = 0; i < 8; ++i) {
      float val = o[j][i] / lrun[i];
      int qr  = qpos + half8 + i;
      int col = h * DHEAD + j * 16 + nloc;
      ctx[((long)(b * SEQ + qr)) * EMB + col] = (bf16)val;
    }
}

// =====================================================================
// Kernel: packed QKV projection. 128x128 tile, 256 thr (8 waves, 4x2),
// double-buffered LDS. V written TRANSPOSED (N,H,D,L).  grid=(64,8,3)
// =====================================================================
__global__ void __launch_bounds__(256, 1)
qkv_proj_kernel(const float* __restrict__ q,
                const float* __restrict__ k,
                const float* __restrict__ v,
                const float* __restrict__ Wp,
                const float* __restrict__ bp,
                bf16* __restrict__ Qw,
                bf16* __restrict__ Kw,
                bf16* __restrict__ Vw) {
  const int z = blockIdx.z;
  const float* X   = (z == 0) ? q : (z == 1) ? k : v;
  const float* W   = Wp + z * (EMB * EMB);
  const float* bia = bp + z * EMB;
  bf16* Out        = (z == 0) ? Qw : (z == 1) ? Kw : Vw;
  const bool vtrans = (z == 2);

  __shared__ alignas(32) bf16 Xs[2][128 * 32];
  __shared__ alignas(32) bf16 Ws[2][128 * 32];

  const int tid  = threadIdx.x;
  const int wave = tid >> 5;
  const int wm   = wave & 3;
  const int wn   = wave >> 2;
  const int lane = tid & 31;
  const int nloc = lane & 15;
  const int half8 = (lane >> 4) * 8;
  const int m0 = blockIdx.x * 128;
  const int n0 = blockIdx.y * 128;

  auto stage = [&](int buf, int k0) {
    for (int i = tid; i < 128 * 32; i += 256) {
      int r = i >> 5, cc = i & 31;
      Xs[buf][i] = (bf16)X[(m0 + r) * EMB + k0 + cc];
      Ws[buf][i] = (bf16)W[(n0 + r) * EMB + k0 + cc];
    }
  };

  v8f c[2][4];
  #pragma unroll
  for (int j = 0; j < 4; ++j) {
    float bv = bia[n0 + wn * 64 + j * 16 + nloc];
    #pragma unroll
    for (int mi = 0; mi < 2; ++mi)
      #pragma unroll
      for (int i = 0; i < 8; ++i) c[mi][j][i] = bv;
  }

  stage(0, 0);
  int buf = 0;
  for (int k0 = 0; k0 < EMB; k0 += 32) {
    __syncthreads();
    if (k0 + 32 < EMB) stage(buf ^ 1, k0 + 32);

    v16bf a0 = load_A_frag(&Xs[buf][(wm * 32) * 32], 32);
    v16bf a1 = load_A_frag(&Xs[buf][(wm * 32 + 16) * 32], 32);
    #pragma unroll
    for (int j = 0; j < 4; ++j) {
      v16bf bfr = load_B_frag(&Ws[buf][(wn * 64 + j * 16) * 32], 32);
      c[0][j] = wmma_bf16(a0, bfr, c[0][j]);
      c[1][j] = wmma_bf16(a1, bfr, c[1][j]);
    }
    buf ^= 1;
  }

  #pragma unroll
  for (int mi = 0; mi < 2; ++mi)
    #pragma unroll
    for (int j = 0; j < 4; ++j)
      #pragma unroll
      for (int i = 0; i < 8; ++i) {
        int m = m0 + wm * 32 + mi * 16 + half8 + i;
        int n = n0 + wn * 64 + j * 16 + nloc;
        int b = m >> 11, l = m & (SEQ - 1);
        int h = n >> 6, d = n & (DHEAD - 1);
        long idx = vtrans
            ? (((long)(b * NHEAD + h) * DHEAD + d) * SEQ + l)   // V: (N,H,D,L)
            : (((long)(b * NHEAD + h) * SEQ + l) * DHEAD + d);  // Q,K: (N,H,L,D)
        Out[idx] = (bf16)c[mi][j][i];
      }
}

// =====================================================================
// Kernel: output projection. Same 128x128 double-buffered tile, f32 out.
// grid=(64,8), block=256
// =====================================================================
__global__ void __launch_bounds__(256, 1)
out_proj_kernel(const bf16* __restrict__ Xc,
                const float* __restrict__ W,
                const float* __restrict__ bia,
                float* __restrict__ out) {
  __shared__ alignas(32) bf16 Xs[2][128 * 32];
  __shared__ alignas(32) bf16 Ws[2][128 * 32];

  const int tid  = threadIdx.x;
  const int wave = tid >> 5;
  const int wm   = wave & 3;
  const int wn   = wave >> 2;
  const int lane = tid & 31;
  const int nloc = lane & 15;
  const int half8 = (lane >> 4) * 8;
  const int m0 = blockIdx.x * 128;
  const int n0 = blockIdx.y * 128;

  auto stage = [&](int buf, int k0) {
    for (int i = tid; i < 128 * 32; i += 256) {
      int r = i >> 5, cc = i & 31;
      Xs[buf][i] = Xc[(m0 + r) * EMB + k0 + cc];
      Ws[buf][i] = (bf16)W[(n0 + r) * EMB + k0 + cc];
    }
  };

  v8f c[2][4];
  #pragma unroll
  for (int j = 0; j < 4; ++j) {
    float bv = bia[n0 + wn * 64 + j * 16 + nloc];
    #pragma unroll
    for (int mi = 0; mi < 2; ++mi)
      #pragma unroll
      for (int i = 0; i < 8; ++i) c[mi][j][i] = bv;
  }

  stage(0, 0);
  int buf = 0;
  for (int k0 = 0; k0 < EMB; k0 += 32) {
    __syncthreads();
    if (k0 + 32 < EMB) stage(buf ^ 1, k0 + 32);

    v16bf a0 = load_A_frag(&Xs[buf][(wm * 32) * 32], 32);
    v16bf a1 = load_A_frag(&Xs[buf][(wm * 32 + 16) * 32], 32);
    #pragma unroll
    for (int j = 0; j < 4; ++j) {
      v16bf bfr = load_B_frag(&Ws[buf][(wn * 64 + j * 16) * 32], 32);
      c[0][j] = wmma_bf16(a0, bfr, c[0][j]);
      c[1][j] = wmma_bf16(a1, bfr, c[1][j]);
    }
    buf ^= 1;
  }

  #pragma unroll
  for (int mi = 0; mi < 2; ++mi)
    #pragma unroll
    for (int j = 0; j < 4; ++j)
      #pragma unroll
      for (int i = 0; i < 8; ++i) {
        int m = m0 + wm * 32 + mi * 16 + half8 + i;
        int n = n0 + wn * 64 + j * 16 + nloc;
        out[(long)m * EMB + n] = c[mi][j][i];
      }
}

// =====================================================================
extern "C" void kernel_launch(void* const* d_in, const int* in_sizes, int n_in,
                              void* d_out, int out_size, void* d_ws, size_t ws_size,
                              hipStream_t stream) {
  const float* query    = (const float*)d_in[0];
  const float* key_t    = (const float*)d_in[1];
  const float* value    = (const float*)d_in[2];
  const float* W_packed = (const float*)d_in[3];
  const float* b_packed = (const float*)d_in[4];
  const float* W_out    = (const float*)d_in[5];
  const float* b_out    = (const float*)d_in[6];
  float* out = (float*)d_out;

  const size_t per = (size_t)NTOK * EMB;
  bf16* Qw  = (bf16*)d_ws;
  bf16* Kw  = Qw + per;
  bf16* Vw  = Kw + per;           // stored transposed (N,H,D,L)
  bf16* ctx = Vw + per;

  qkv_proj_kernel<<<dim3(NTOK / 128, EMB / 128, 3), dim3(256), 0, stream>>>(
      query, key_t, value, W_packed, b_packed, Qw, Kw, Vw);
  attn_kernel<<<dim3(SEQ / 64, NBATCH * NHEAD), dim3(128), 0, stream>>>(
      Qw, Kw, Vw, ctx);
  out_proj_kernel<<<dim3(NTOK / 128, EMB / 128), dim3(256), 0, stream>>>(
      ctx, W_out, b_out, out);
}
